// IRAgent_76484777607573
// MI455X (gfx1250) — matrix operand order repaired and checked by reading, not compiled, not run on hardware
//
#include <hip/hip_runtime.h>
#include <hip/hip_bf16.h>
#include <math.h>

// ---------------------------------------------------------------------------
// IRAgent fused forward, MI455X (gfx1250, wave32, WMMA 16x16x32 f16).
// One workgroup = 128 threads (4 waves) = 8 agents. Everything fused, all
// intermediates in LDS, all linear layers via v_wmma_f32_16x16x32_f16.
// ---------------------------------------------------------------------------

typedef _Float16 h16;
typedef __attribute__((ext_vector_type(16))) _Float16 v16h;
typedef __attribute__((ext_vector_type(8)))  float    v8f;

#define THREADS 128
#define AG      8                 // agents per block
#define ROWS_E  (AG * 10 * 2)     // 160 enemy rows (agent,e,k)
#define ROWS_A  (AG * 9 * 2)      // 144 ally rows
#define ROWS    (ROWS_E + ROWS_A) // 304

union AF { v16h v; unsigned int u[8]; };

__device__ __forceinline__ v8f zero8() {
  v8f z;
#pragma unroll
  for (int i = 0; i < 8; ++i) z[i] = 0.f;
  return z;
}

__device__ __forceinline__ v8f wmma32(v16h a, v16h b, v8f c) {
  return __builtin_amdgcn_wmma_f32_16x16x32_f16(false, a, false, b,
                                                (short)0, c, false, false);
}

// A fragment (16x32, f16) from row-major LDS: lane layout per ISA 7.12.2.
__device__ __forceinline__ v16h load_a(const h16* base, int lda) {
  int l = threadIdx.x & 31;
  const h16* row = base + (l & 15) * lda;
  int kh = (l >> 4) * 8;
  AF f;
#pragma unroll
  for (int v = 0; v < 8; ++v) {
    int k = ((v & 4) ? 16 : 0) + kh + 2 * (v & 3);
    f.u[v] = *(const unsigned int*)(row + k);
  }
  return f.v;
}

// A fragment with per-row source gather (for concat inputs).
template <typename RowFn>
__device__ __forceinline__ v16h load_a_fn(RowFn rowp) {
  int l = threadIdx.x & 31;
  const h16* row = rowp(l & 15);
  int kh = (l >> 4) * 8;
  AF f;
#pragma unroll
  for (int v = 0; v < 8; ++v) {
    int k = ((v & 4) ? 16 : 0) + kh + 2 * (v & 3);
    f.u[v] = *(const unsigned int*)(row + k);
  }
  return f.v;
}

// B fragment (32x16) from transposed weights Wt[n*K + k].
__device__ __forceinline__ v16h load_b(const h16* wt, int K, int nt, int kt) {
  int l = threadIdx.x & 31;
  const h16* row = wt + ((l & 15) + nt * 16) * K + kt * 32;
  int kh = (l >> 4) * 8;
  AF f;
#pragma unroll
  for (int v = 0; v < 8; ++v)
    f.u[v] = *(const unsigned int*)(row + ((v & 4) ? 16 : 0) + kh + 2 * (v & 3));
  return f.v;
}

// Store 16x16 f32 accumulator to f16 row-major LDS. No bias path has NO adds.
template <bool RELU>
__device__ __forceinline__ void store_c(h16* dst, int ldd, v8f acc) {
  int l = threadIdx.x & 31;
  int n = l & 15;
#pragma unroll
  for (int v = 0; v < 8; ++v) {
    float x = acc[v];
    if (RELU) x = fmaxf(x, 0.f);
    dst[(v + 8 * (l >> 4)) * ldd + n] = (h16)x;
  }
}

__device__ __forceinline__ void store_c_br(h16* dst, int ldd, v8f acc,
                                           const float* bias) {
  int l = threadIdx.x & 31;
  int n = l & 15;
  float b = bias[n];
#pragma unroll
  for (int v = 0; v < 8; ++v) {
    float x = fmaxf(acc[v] + b, 0.f);
    dst[(v + 8 * (l >> 4)) * ldd + n] = (h16)x;
  }
}

// 64-wide dot of staged row (ld 128) with f32 weights; lanes split the K range
// (lanes 0-15 take K 0..31 of row l, lanes 16-31 take K 32..63), combined with
// a cross-half shuffle. Result valid in all lanes; row = lane & 15.
__device__ __forceinline__ float dot64_stg(const h16* stg, const float* w) {
  int l = threadIdx.x & 31;
  int rl = l & 15, hb = (l >> 4) * 32;
  float s = 0.f;
#pragma unroll
  for (int hh = 0; hh < 32; ++hh)
    s += (float)stg[rl * 128 + hb + hh] * w[hb + hh];
  s += __shfl_xor(s, 16, 32);
  return s;
}

// Cooperative: load W (fi x fo, row-major f32) transposed into dst[fo][Kpad] f16.
__device__ __forceinline__ void load_wt(h16* dst, const float* W,
                                        int fi, int fo, int Kpad) {
  for (int idx = threadIdx.x; idx < fo * Kpad; idx += THREADS) {
    int n = idx / Kpad, k = idx - n * Kpad;
    dst[idx] = (k < fi) ? (h16)W[k * fo + n] : (h16)0.f;
  }
}

__device__ __forceinline__ void load_f32(float* dst, const float* src, int n) {
  for (int i = threadIdx.x; i < n; i += THREADS) dst[i] = src[i];
}

struct P {
  const float *own, *enemy, *ally;
  const float *rcnE0, *rcnE1, *rcnA0, *rcnA1, *rcnF0, *rcnF1;
  const float *irE0, *irE1, *irE2, *irA0, *irA1, *irA2;
  const float *atk0, *atk1, *atk2, *atkB0, *atkB1, *atkB2;
  const float *oth0, *oth1, *othB0, *othB1;
  float* out;
};

__global__ __launch_bounds__(THREADS, 1) void iragent_kernel(P p) {
  __shared__ h16   sVec[ROWS * 32];        // vec-features, K padded 19->32
  __shared__ h16   sRC[ROWS * 64];         // e_rc / a_rc (later reused as attack h1)
  __shared__ h16   sWt[128 * 128];         // transposed weight staging
  __shared__ h16   sStg[4 * 16 * 128];     // per-wave activation staging
  __shared__ h16   sEsc[AG * 10 * 64];     // e_scalar
  __shared__ h16   sPooled[AG * 2 * 64];
  __shared__ h16   sVecOut[AG * 2 * 64];   // vec after rcn_feat
  __shared__ h16   sScalH[16 * 64];        // scal as f16, rows 8..15 zero
  __shared__ h16   sOthH[16 * 64];         // other-net hidden (f16)
  __shared__ float sScal[AG * 64];
  __shared__ float sImp[ROWS_E];           // e_imp + a_imp per (agent,e,k)
  __shared__ float sMove[AG * 2];
  __shared__ float sBias[192];
  __shared__ float sW2[64];                // final-layer dot weights
  __shared__ float sOtherQ[AG * 3];
  __shared__ float sAtkQ[AG * 10];

  const int tid  = threadIdx.x;
  const int wave = tid >> 5;
  h16* stg = sStg + wave * (16 * 128);
  const long gA0 = (long)blockIdx.x * AG;

  // ---- Stage 0: rcn weights + vec-features --------------------------------
  load_wt(sWt + 0,    p.rcnE0, 19, 64, 32);
  load_wt(sWt + 2048, p.rcnE1, 64, 64, 64);
  load_wt(sWt + 6144, p.rcnA0, 19, 64, 32);
  load_wt(sWt + 8192, p.rcnA1, 64, 64, 64);

  for (int r = tid; r < ROWS; r += THREADS) {
    int agent, ent, kd; const float* f;
    if (r < ROWS_E) {
      agent = r / 20; int rem = r - agent * 20; ent = rem >> 1; kd = rem & 1;
      f = p.enemy + ((gA0 + agent) * 10 + ent) * 9;
    } else {
      int r2 = r - ROWS_E;
      agent = r2 / 18; int rem = r2 - agent * 18; ent = rem >> 1; kd = rem & 1;
      f = p.ally + ((gA0 + agent) * 9 + ent) * 9;
    }
    const float* ow = p.own + (gA0 + agent) * 11;
    float f2 = f[2], f3 = f[3];
    float pm = sqrtf(f2 * f2 + f3 * f3);
    float inv = (pm == 0.f) ? 0.f : (1.41421356237f / pm); // sqrt2/|pos|, 0-guard
    float pk = kd ? f3 : f2;
    h16* row = sVec + r * 32;
    row[0] = (h16)pk;
#pragma unroll
    for (int j = 0; j < 7; ++j) {
      float o = (j < 2) ? f[j] : f[j + 2];
      row[1 + j] = (h16)(pk * fabsf(o) * inv);
    }
#pragma unroll
    for (int j = 0; j < 11; ++j) row[8 + j] = (h16)(pk * fabsf(ow[j]) * inv);
#pragma unroll
    for (int j = 19; j < 32; ++j) row[j] = (h16)0.f;
  }
  __syncthreads();

  // ---- Stage 1: rcn_enemy / rcn_ally (19 m-tiles, wave-striped) -----------
  for (int t = wave; t < 19; t += 4) {
    bool ally = t >= 10;
    int mrow0 = ally ? ROWS_E + (t - 10) * 16 : t * 16;
    const h16* w1 = sWt + (ally ? 6144 : 0);
    const h16* w2 = sWt + (ally ? 8192 : 2048);
    v16h a = load_a(sVec + mrow0 * 32, 32);
#pragma unroll
    for (int nt = 0; nt < 4; ++nt) {
      v8f acc = zero8();
      acc = wmma32(a, load_b(w1, 32, nt, 0), acc);
      store_c<true>(stg + nt * 16, 128, acc);               // relu
    }
    v16h a0 = load_a(stg, 128);
    v16h a1 = load_a(stg + 32, 128);
#pragma unroll
    for (int nt = 0; nt < 4; ++nt) {
      v8f acc = zero8();
      acc = wmma32(a0, load_b(w2, 64, nt, 0), acc);
      acc = wmma32(a1, load_b(w2, 64, nt, 1), acc);
      store_c<false>(sRC + mrow0 * 64 + nt * 16, 64, acc);  // no relu
    }
  }
  __syncthreads();

  // ---- Stage 2: e_scalar + quaternion max-pool; stage rcn_feat weights ----
  load_wt(sWt + 0,    p.rcnF0, 64, 64, 64);
  load_wt(sWt + 4096, p.rcnF1, 64, 64, 64);
  for (int idx = tid; idx < AG * 10 * 64; idx += THREADS) {
    int ae = idx >> 6, ch = idx & 63;
    float x0 = (float)sRC[(ae * 2) * 64 + ch];
    float x1 = (float)sRC[(ae * 2 + 1) * 64 + ch];
    sEsc[idx] = (h16)(x0 * x0 + x1 * x1);
  }
  for (int idx = tid; idx < AG * 64; idx += THREADS) {
    int a = idx >> 6, ch = idx & 63;
    float best = -1.f, b0 = 0.f, b1 = 0.f;
    for (int ent = 0; ent < 19; ++ent) {
      int rb = (ent < 10) ? (a * 20 + ent * 2)
                          : (ROWS_E + a * 18 + (ent - 10) * 2);
      float x0 = (float)sRC[rb * 64 + ch];
      float x1 = (float)sRC[(rb + 1) * 64 + ch];
      float m = x0 * x0 + x1 * x1;
      if (m > best) { best = m; b0 = x0; b1 = x1; }   // first-max like argmax
    }
    sPooled[(a * 2) * 64 + ch] = (h16)b0;
    sPooled[(a * 2 + 1) * 64 + ch] = (h16)b1;
  }
  __syncthreads();

  // ---- Stage 3: rcn_feat (one 16-row tile, wave 0) ------------------------
  if (wave == 0) {
    v16h a0 = load_a(sPooled, 64);
    v16h a1 = load_a(sPooled + 32, 64);
#pragma unroll
    for (int nt = 0; nt < 4; ++nt) {
      v8f acc = zero8();
      acc = wmma32(a0, load_b(sWt, 64, nt, 0), acc);
      acc = wmma32(a1, load_b(sWt, 64, nt, 1), acc);
      store_c<true>(stg + nt * 16, 128, acc);
    }
    v16h h0 = load_a(stg, 128);
    v16h h1 = load_a(stg + 32, 128);
#pragma unroll
    for (int nt = 0; nt < 4; ++nt) {
      v8f acc = zero8();
      acc = wmma32(h0, load_b(sWt + 4096, 64, nt, 0), acc);
      acc = wmma32(h1, load_b(sWt + 4096, 64, nt, 1), acc);
      store_c<false>(sVecOut + nt * 16, 64, acc);
    }
  }
  __syncthreads();

  // ---- Stage 4: scal; stage ir_enemy + other-net weights ------------------
  load_wt(sWt + 0,     p.irE0, 128, 64, 128);
  load_wt(sWt + 8192,  p.irE1, 64, 64, 64);
  load_wt(sWt + 12288, p.oth0, 64, 64, 64);   // other-net layer1 (fits exactly)
  load_f32(sW2, p.irE2, 64);
  load_f32(sBias + 128, p.othB0, 64);
  for (int idx = tid; idx < 16 * 64; idx += THREADS) {
    int a = idx >> 6, ch = idx & 63;
    float s = 0.f;
    if (a < AG) {
      float v0 = (float)sVecOut[(a * 2) * 64 + ch];
      float v1 = (float)sVecOut[(a * 2 + 1) * 64 + ch];
      s = v0 * v0 + v1 * v1;
      sScal[idx] = s;
    }
    sScalH[idx] = (h16)s;                     // rows 8..15 zero-padded
  }
  __syncthreads();

  // ---- Stage 5: ir_enemy then ir_ally (same input, summed) ----------------
  for (int net = 0; net < 2; ++net) {
    for (int t = wave; t < 10; t += 4) {
      int mrow0 = t * 16;
      v16h af[4];
#pragma unroll
      for (int kt = 0; kt < 4; ++kt) {
        const h16* vecO = sVecOut; const h16* rc = sRC;
        af[kt] = load_a_fn([=](int m) -> const h16* {
          int r = mrow0 + m;
          int agent = r / 20, kd = r & 1;
          return (kt < 2) ? (vecO + (agent * 2 + kd) * 64 + kt * 32)
                          : (rc + r * 64 + (kt - 2) * 32);
        });
      }
#pragma unroll
      for (int nt = 0; nt < 4; ++nt) {
        v8f acc = zero8();
#pragma unroll
        for (int kt = 0; kt < 4; ++kt)
          acc = wmma32(af[kt], load_b(sWt, 128, nt, kt), acc);
        store_c<true>(stg + nt * 16, 128, acc);
      }
      v16h h0 = load_a(stg, 128);
      v16h h1 = load_a(stg + 32, 128);
      v8f acc2[4];
#pragma unroll
      for (int nt = 0; nt < 4; ++nt) {
        acc2[nt] = zero8();
        acc2[nt] = wmma32(h0, load_b(sWt + 8192, 64, nt, 0), acc2[nt]);
        acc2[nt] = wmma32(h1, load_b(sWt + 8192, 64, nt, 1), acc2[nt]);
      }
#pragma unroll
      for (int nt = 0; nt < 4; ++nt)
        store_c<true>(stg + nt * 16, 128, acc2[nt]);
      float s = dot64_stg(stg, sW2);          // final 64->1 dot (no relu)
      int l = tid & 31;
      if (l < 16) {
        int r = mrow0 + l;
        if (net == 0) sImp[r] = s; else sImp[r] += s;
      }
    }
    if (net == 0 && wave == 3) {              // other-net layer1 via WMMA
      v16h a0 = load_a(sScalH, 64);
      v16h a1 = load_a(sScalH + 32, 64);
#pragma unroll
      for (int nt = 0; nt < 4; ++nt) {
        v8f acc = zero8();
        acc = wmma32(a0, load_b(sWt + 12288, 64, nt, 0), acc);
        acc = wmma32(a1, load_b(sWt + 12288, 64, nt, 1), acc);
        store_c_br(sOthH + nt * 16, 64, acc, sBias + 128 + nt * 16);
      }
    }
    __syncthreads();
    if (net == 0) {
      load_wt(sWt + 0,    p.irA0, 128, 64, 128);
      load_wt(sWt + 8192, p.irA1, 64, 64, 64);
      load_f32(sW2, p.irA2, 64);
      __syncthreads();
    }
  }

  // ---- Stage 6: move_vec; stage attack layer-1 weights --------------------
  if (tid < AG * 2) {
    int a = tid >> 1, kd = tid & 1;
    float s = 0.f;
    for (int e = 0; e < 10; ++e) s += sImp[a * 20 + e * 2 + kd];
    sMove[tid] = s;
  }
  load_wt(sWt, p.atk0, 128, 128, 128);
  load_f32(sBias, p.atkB0, 128);
  load_f32(sW2, p.atk2, 64);
  __syncthreads();

  // ---- Stage 7: attack layer1 (e_rc dead -> reuse sRC as h1[80][128]) -----
  h16* hAtk = sRC;
  for (int t = wave; t < 5; t += 4) {
    int mrow0 = t * 16;
    v16h af[4];
#pragma unroll
    for (int kt = 0; kt < 4; ++kt) {
      const h16* sc = sScalH; const h16* esc = sEsc;
      af[kt] = load_a_fn([=](int m) -> const h16* {
        int r = mrow0 + m;
        int agent = r / 10;
        return (kt < 2) ? (sc + agent * 64 + kt * 32)
                        : (esc + r * 64 + (kt - 2) * 32);
      });
    }
#pragma unroll
    for (int nt = 0; nt < 8; ++nt) {
      v8f acc = zero8();
#pragma unroll
      for (int kt = 0; kt < 4; ++kt)
        acc = wmma32(af[kt], load_b(sWt, 128, nt, kt), acc);
      store_c_br(hAtk + mrow0 * 128 + nt * 16, 128, acc, sBias + nt * 16);
    }
  }
  if (tid < AG * 3) {                       // other-net layer2 (tiny)
    int a = tid / 3, o = tid - a * 3;
    float s = p.othB1[o];
#pragma unroll
    for (int hh = 0; hh < 64; ++hh)
      s += (float)sOthH[a * 64 + hh] * p.oth1[hh * 3 + o];
    sOtherQ[tid] = s;
  }
  __syncthreads();

  // ---- Stage 8: attack layer2 + final dot ---------------------------------
  load_wt(sWt, p.atk1, 128, 64, 128);
  load_f32(sBias, p.atkB1, 64);
  __syncthreads();
  float b2 = p.atkB2[0];
  for (int t = wave; t < 5; t += 4) {
    int mrow0 = t * 16;
    v16h a[4];
#pragma unroll
    for (int kt = 0; kt < 4; ++kt) a[kt] = load_a(hAtk + mrow0 * 128 + kt * 32, 128);
    v8f acc2[4];
#pragma unroll
    for (int nt = 0; nt < 4; ++nt) {
      acc2[nt] = zero8();
#pragma unroll
      for (int kt = 0; kt < 4; ++kt)
        acc2[nt] = wmma32(a[kt], load_b(sWt, 128, nt, kt), acc2[nt]);
    }
#pragma unroll
    for (int nt = 0; nt < 4; ++nt)
      store_c_br(stg + nt * 16, 128, acc2[nt], sBias + nt * 16);
    float s = b2 + dot64_stg(stg, sW2);
    int l = tid & 31;
    if (l < 16) sAtkQ[mrow0 + l] = s;
  }
  __syncthreads();

  // ---- Stage 9: assemble output [other_q(2) | up,down,right,left | attack(10)]
  {
    int a = tid >> 4, j = tid & 15;
    float val;
    if (j < 2) val = sOtherQ[a * 3 + j];
    else if (j < 6) {
      float mm  = sOtherQ[a * 3 + 2];
      float mv0 = sMove[a * 2 + 0], mv1 = sMove[a * 2 + 1];
      val = (j == 2) ? mm + mv1 : (j == 3) ? mm - mv1
          : (j == 4) ? mm + mv0 : mm - mv0;
    } else {
      val = sAtkQ[a * 10 + (j - 6)];
    }
    p.out[(gA0 + a) * 16 + j] = val;
  }
}

extern "C" void kernel_launch(void* const* d_in, const int* in_sizes, int n_in,
                              void* d_out, int out_size, void* d_ws, size_t ws_size,
                              hipStream_t stream) {
  (void)n_in; (void)out_size; (void)d_ws; (void)ws_size;
  P p;
  // Feature tensors: insertion order (own,enemy,ally) or sorted (ally,enemy,own).
  if (in_sizes[0] == 2048 * 10 * 11) {
    p.own = (const float*)d_in[0]; p.enemy = (const float*)d_in[1];
    p.ally = (const float*)d_in[2];
  } else {
    p.ally = (const float*)d_in[0]; p.enemy = (const float*)d_in[1];
    p.own = (const float*)d_in[2];
  }
  const float** w = (const float**)(d_in + 3);
  if (in_sizes[3] == 19 * 64) {
    // params flattened in dict-insertion order
    p.rcnE0 = w[0];  p.rcnE1 = w[1];
    p.rcnA0 = w[2];  p.rcnA1 = w[3];
    p.rcnF0 = w[4];  p.rcnF1 = w[5];
    p.irE0  = w[6];  p.irE1  = w[7];  p.irE2 = w[8];
    p.irA0  = w[9];  p.irA1  = w[10]; p.irA2 = w[11];
    p.atk0  = w[12]; p.atk1  = w[13]; p.atk2 = w[14];
    p.atkB0 = w[15]; p.atkB1 = w[16]; p.atkB2 = w[17];
    p.oth0  = w[18]; p.oth1  = w[19];
    p.othB0 = w[20]; p.othB1 = w[21];
  } else {
    // params flattened with sorted dict keys (jax tree order)
    p.atkB0 = w[0];  p.atkB1 = w[1];  p.atkB2 = w[2];
    p.atk0  = w[3];  p.atk1  = w[4];  p.atk2  = w[5];
    p.irA0  = w[6];  p.irA1  = w[7];  p.irA2  = w[8];
    p.irE0  = w[9];  p.irE1  = w[10]; p.irE2  = w[11];
    p.othB0 = w[12]; p.othB1 = w[13];
    p.oth0  = w[14]; p.oth1  = w[15];
    p.rcnA0 = w[16]; p.rcnA1 = w[17];
    p.rcnE0 = w[18]; p.rcnE1 = w[19];
    p.rcnF0 = w[20]; p.rcnF1 = w[21];
  }
  p.out = (float*)d_out;

  dim3 grid((2048 * 10) / AG);   // 2560 blocks
  dim3 block(THREADS);
  hipLaunchKernelGGL(iragent_kernel, grid, block, 0, stream, p);
}
